// VectorQuantizer_34505767256300
// MI455X (gfx1250) — matrix operand order repaired and checked by reading, not compiled, not run on hardware
//
#include <hip/hip_runtime.h>
#include <hip/hip_bf16.h>

#define NROWS 32768
#define KCODES 4096
#define DIM 512
#define NCH 16            // DIM / 32 k-chunks per dot product
#define NTILES (KCODES/16)

typedef __attribute__((ext_vector_type(16))) __bf16 v16bf;
typedef __attribute__((ext_vector_type(8)))  float  v8f;

union BF16x16 {
  v16bf v;
  unsigned short s[16];
  uint4 q[2];
};

__device__ __forceinline__ unsigned short f32_to_bf16u(float f) {
  union { float f; unsigned u; } v; v.f = f;
  unsigned r = v.u + 0x7FFFu + ((v.u >> 16) & 1u);   // round-to-nearest-even
  return (unsigned short)(r >> 16);
}

__device__ __forceinline__ float bf16u_to_f32(unsigned short h) {
  union { unsigned u; float f; } v; v.u = ((unsigned)h) << 16;
  return v.f;
}

// ---------------------------------------------------------------------------
// Prep: codebook fp32 -> bf16 hi/lo planes + per-code squared norm.
// grid = KCODES blocks, 128 threads.
// ---------------------------------------------------------------------------
__global__ __launch_bounds__(128)
void vq_prep_codebook(const float* __restrict__ cb,
                      unsigned short* __restrict__ cb_hi,
                      unsigned short* __restrict__ cb_lo,
                      float* __restrict__ cnorm) {
  const int k = blockIdx.x;
  const int t = threadIdx.x;
  float s = 0.0f;
  for (int i = t; i < DIM; i += 128) {
    float v = cb[(size_t)k * DIM + i];
    unsigned short h = f32_to_bf16u(v);
    float fh = bf16u_to_f32(h);
    unsigned short l = f32_to_bf16u(v - fh);
    cb_hi[(size_t)k * DIM + i] = h;
    cb_lo[(size_t)k * DIM + i] = l;
    s += v * v;
  }
  #pragma unroll
  for (int off = 16; off >= 1; off >>= 1)
    s += __shfl_xor(s, off, 32);
  __shared__ float red[4];
  if ((t & 31) == 0) red[t >> 5] = s;
  __syncthreads();
  if (t == 0) cnorm[k] = red[0] + red[1] + red[2] + red[3];
}

// ---------------------------------------------------------------------------
// Main: per wave, 16 rows of x vs all 4096 codes via bf16 hi/lo WMMA.
// block = 128 threads (4 waves) -> 64 rows / block; grid = NROWS/64.
// ---------------------------------------------------------------------------
__global__ __launch_bounds__(128)
void vq_argmin_kernel(const float* __restrict__ x,
                      const unsigned short* __restrict__ cb_hi,
                      const unsigned short* __restrict__ cb_lo,
                      const float* __restrict__ cnorm,
                      int* __restrict__ ids) {
  const int lane = threadIdx.x & 31;
  const int wave = threadIdx.x >> 5;
  const int row0 = (blockIdx.x * 4 + wave) * 16;
  const int lrow = lane & 15;   // A row within strip / B column within tile
  const int grp  = lane >> 4;   // lane half selects K sub-range per ISA layout

  // -------- Preload A strip (16 rows x 512 feats) as bf16 hi/lo fragments.
  // A 16x32 bf16 layout: lane&15 = row M; lanes 0-15 hold K 0..7,16..23;
  // lanes 16-31 hold K 8..15,24..31 (packed 2/VGPR).
  v16bf a_hi[NCH], a_lo[NCH];
  const float* xrow = x + (size_t)(row0 + lrow) * DIM;
  #pragma unroll
  for (int ch = 0; ch < NCH; ++ch) {
    BF16x16 h, l;
    const int f0 = ch * 32 + grp * 8;
    #pragma unroll
    for (int e = 0; e < 8; ++e) {
      float v0 = xrow[f0 + e];
      float v1 = xrow[f0 + 16 + e];
      unsigned short h0 = f32_to_bf16u(v0);
      unsigned short h1 = f32_to_bf16u(v1);
      h.s[e]     = h0;
      h.s[8 + e] = h1;
      l.s[e]     = f32_to_bf16u(v0 - bf16u_to_f32(h0));
      l.s[8 + e] = f32_to_bf16u(v1 - bf16u_to_f32(h1));
    }
    a_hi[ch] = h.v;
    a_lo[ch] = l.v;
  }

  float minv[8];
  int   mini[8];
  #pragma unroll
  for (int j = 0; j < 8; ++j) { minv[j] = 3.4e38f; mini[j] = 0; }

  const int bgrp = grp * 8;     // B 32x16 layout: lane half holds K 8h..8h+7, 16+8h..16+8h+7

  for (int t = 0; t < NTILES; ++t) {
    const int code = t * 16 + lrow;
    const size_t brow = (size_t)code * DIM;
    const float cn = cnorm[code];

    // Prefetch next tile's B rows (CDNA5 global_prefetch path).
    if (t + 1 < NTILES) {
      __builtin_prefetch(cb_hi + brow + (size_t)16 * DIM, 0, 3);
      __builtin_prefetch(cb_lo + brow + (size_t)16 * DIM, 0, 3);
    }

    v8f acc = {};
    #pragma unroll
    for (int ch = 0; ch < NCH; ++ch) {
      BF16x16 bh, bl;
      const uint4* ph = (const uint4*)(cb_hi + brow + ch * 32 + bgrp);
      const uint4* pl = (const uint4*)(cb_lo + brow + ch * 32 + bgrp);
      bh.q[0] = ph[0]; bh.q[1] = ph[2];
      bl.q[0] = pl[0]; bl.q[1] = pl[2];
      // dot ~= hi*hi + hi*lo + lo*hi  (fp32-grade accuracy for argmin)
      acc = __builtin_amdgcn_wmma_f32_16x16x32_bf16(false, a_hi[ch], false, bh.v,
                                                    (short)0, acc, false, false);
      acc = __builtin_amdgcn_wmma_f32_16x16x32_bf16(false, a_hi[ch], false, bl.v,
                                                    (short)0, acc, false, false);
      acc = __builtin_amdgcn_wmma_f32_16x16x32_bf16(false, a_lo[ch], false, bh.v,
                                                    (short)0, acc, false, false);
    }

    // D layout: VGPR j, lane -> row M = j + 8*grp, col N = lane&15 = this code.
    #pragma unroll
    for (int j = 0; j < 8; ++j) {
      float d = cn - 2.0f * acc[j];
      if (d < minv[j]) { minv[j] = d; mini[j] = code; }
    }
  }

  // Cross-lane argmin within each 16-lane column group (first-index tiebreak).
  #pragma unroll
  for (int j = 0; j < 8; ++j) {
    float v = minv[j];
    int   i = mini[j];
    #pragma unroll
    for (int off = 8; off >= 1; off >>= 1) {
      float ov = __shfl_xor(v, off, 32);
      int   oi = __shfl_xor(i, off, 32);
      if (ov < v || (ov == v && oi < i)) { v = ov; i = oi; }
    }
    if (lrow == 0) ids[row0 + grp * 8 + j] = i;
  }
}

// ---------------------------------------------------------------------------
// Gather: z_q[n] = codebook[ids[n]]  (one block per row, float4 copies).
// ---------------------------------------------------------------------------
__global__ __launch_bounds__(128)
void vq_gather_kernel(const float* __restrict__ cb,
                      const int* __restrict__ ids,
                      float* __restrict__ zq) {
  const int row = blockIdx.x;
  const int id = ids[row];
  const float4* src = (const float4*)(cb + (size_t)id * DIM);
  float4* dst = (float4*)(zq + (size_t)row * DIM);
  dst[threadIdx.x] = src[threadIdx.x];   // 128 * 16B = 2048B = DIM floats
}

extern "C" void kernel_launch(void* const* d_in, const int* in_sizes, int n_in,
                              void* d_out, int out_size, void* d_ws, size_t ws_size,
                              hipStream_t stream) {
  const float* x  = (const float*)d_in[0];   // [NROWS, DIM] fp32
  const float* cb = (const float*)d_in[1];   // [KCODES, DIM] fp32

  float* zq  = (float*)d_out;                          // [NROWS, DIM] fp32
  int*   ids = (int*)((float*)d_out + (size_t)NROWS * DIM);  // [NROWS] int32

  unsigned short* cb_hi = (unsigned short*)d_ws;                  // 4 MB
  unsigned short* cb_lo = cb_hi + (size_t)KCODES * DIM;           // 4 MB
  float*          cnorm = (float*)(cb_lo + (size_t)KCODES * DIM); // 16 KB

  vq_prep_codebook<<<KCODES, 128, 0, stream>>>(cb, cb_hi, cb_lo, cnorm);
  vq_argmin_kernel<<<NROWS / 64, 128, 0, stream>>>(x, cb_hi, cb_lo, cnorm, ids);
  vq_gather_kernel<<<NROWS, 128, 0, stream>>>(cb, ids, zq);
}
